// MLA_66649302499498
// MI455X (gfx1250) — compile-verified, matrix operand order checked by reading, not run on hardware
//
#include <hip/hip_runtime.h>

// ---------------- problem constants ----------------
#define B_    2
#define S_    2048
#define D_    2048
#define H_    16
#define NOPE_ 128
#define ROPE_ 64
#define QKD_  192          // NOPE + ROPE
#define VD_   128
#define MTOT  (B_ * S_)    // 4096
#define NQK   (H_ * QKD_)  // 3072
#define NV    (H_ * VD_)   // 2048

typedef unsigned short ushort_t;
typedef __attribute__((ext_vector_type(16))) __bf16 v16bf;
typedef __attribute__((ext_vector_type(8)))  __bf16 v8bf;
typedef __attribute__((ext_vector_type(8)))  float  v8f;

// ---------------- bf16 helpers (bit-exact RNE) ----------------
static __device__ __forceinline__ ushort_t f2bf_bits(float f) {
    union { float f; unsigned u; } x; x.f = f;
    unsigned r = x.u + 0x7FFFu + ((x.u >> 16) & 1u);
    return (ushort_t)(r >> 16);
}
static __device__ __forceinline__ float bf2f(ushort_t u) {
    union { unsigned u; float f; } x; x.u = ((unsigned)u) << 16;
    return x.f;
}
static __device__ __forceinline__ __bf16 bits2bf(ushort_t u) {
    union { ushort_t u; __bf16 b; } x; x.u = u; return x.b;
}

// Load one 16-lane-row fragment in the CDNA5 16-bit A/B VGPR layout.
// p points at (row base) + kbase + 8*g  (g = lane>>4).
// elements 0..7  <- K = kbase + 8g .. +7   (contiguous, 16B)
// elements 8..15 <- K = kbase + 16 + 8g .. (contiguous, 16B)
static __device__ __forceinline__ v16bf load_frag(const ushort_t* p) {
    const v8bf lo = *reinterpret_cast<const v8bf*>(p);
    const v8bf hi = *reinterpret_cast<const v8bf*>(p + 16);
    return __builtin_shufflevector(lo, hi, 0,1,2,3,4,5,6,7,8,9,10,11,12,13,14,15);
}

static __device__ __forceinline__ v8f wmma_bf16(v16bf a, v16bf b, v8f c) {
    return __builtin_amdgcn_wmma_f32_16x16x32_bf16(
        /*neg_a=*/false, a, /*neg_b=*/false, b,
        /*c_mod=*/(short)0, c, /*reuse_a=*/false, /*reuse_b=*/false);
}

// ---------------- async global->LDS copy (CDNA5 path, guarded) ----------------
#if __has_builtin(__builtin_amdgcn_global_load_async_to_lds_b128)
#define HAVE_ASYNC_LDS 1
typedef int v4i __attribute__((vector_size(16)));
typedef __attribute__((address_space(1))) v4i* g_v4i_p;
typedef __attribute__((address_space(3))) v4i* l_v4i_p;
static __device__ __forceinline__ void cp16_g2l(const ushort_t* g, ushort_t* l) {
    // per-lane: LDS[l] = MEM[g], 16 bytes; tracked by ASYNCcnt
    __builtin_amdgcn_global_load_async_to_lds_b128(
        (g_v4i_p)(v4i*)g, (l_v4i_p)(v4i*)l, 0, 0);
}
static __device__ __forceinline__ void wait_async() {
    asm volatile("s_wait_asynccnt 0" ::: "memory");
}
#else
static __device__ __forceinline__ void cp16_g2l(const ushort_t* g, ushort_t* l) {
    *reinterpret_cast<uint4*>(l) = *reinterpret_cast<const uint4*>(g);
}
static __device__ __forceinline__ void wait_async() {}
#endif

// ---------------- conversion kernels ----------------
__global__ void conv_bf16(const float* __restrict__ X, ushort_t* __restrict__ Y, long total) {
    long i = (long)blockIdx.x * blockDim.x + threadIdx.x;
    if (i < total) Y[i] = f2bf_bits(X[i]);
}

// W: [K][N] f32 row-major  ->  WT: [N][K] bf16 row-major
__global__ void conv_transpose(const float* __restrict__ W, ushort_t* __restrict__ WT,
                               int K, int N) {
    long i = (long)blockIdx.x * blockDim.x + threadIdx.x;
    long total = (long)K * N;
    if (i >= total) return;
    int k = (int)(i % K);
    int n = (int)(i / K);
    WT[i] = f2bf_bits(W[(long)k * N + n]);
}

// ---------------- WMMA GEMM: C[M][N] = A[M][K](bf16) * BT[N][K]^T + bias ----------------
// block = 256 threads (8 waves as 2(M) x 4(N)); block tile 64(M) x 256(N);
// wave tile 32 x 64 (8 accumulators). A/B K-slabs double-buffered in LDS via
// async global->LDS copies.
template <bool OUT_BF16>
__global__ void __launch_bounds__(256)
gemm_wmma(const ushort_t* __restrict__ A, const ushort_t* __restrict__ BT,
          const float* __restrict__ bias, void* __restrict__ Cout,
          int M, int N, int K) {
    __shared__ __align__(16) ushort_t As[2][64 * 32];    //  8 KB
    __shared__ __align__(16) ushort_t Bs[2][256 * 32];   // 32 KB

    const int tid  = threadIdx.x;
    const int lane = tid & 31;
    const int wave = tid >> 5;
    const int mw   = wave >> 2;          // 0..1
    const int nw   = wave & 3;           // 0..3
    const int g    = lane >> 4;
    const int r16  = lane & 15;

    const int mBlk = blockIdx.y * 64;    // block M origin
    const int nBlk = blockIdx.x * 256;   // block N origin
    const int m0   = mBlk + mw * 32;     // wave M origin
    const int n0   = nBlk + nw * 64;     // wave N origin

    const int sRow = tid >> 2;           // 0..63  (staging row)
    const int sSeg = tid & 3;            // 0..3   (16B segment within 64B row)

    // stage one K-slab (A: 64x32, B: 256x32) into LDS buffer `bi`
    auto stage = [&](int bi, int k0) {
        cp16_g2l(A + (size_t)(mBlk + sRow) * K + k0 + sSeg * 8,
                 &As[bi][sRow * 32 + sSeg * 8]);
        #pragma unroll
        for (int j = 0; j < 4; ++j) {
            int br = j * 64 + sRow;
            cp16_g2l(BT + (size_t)(nBlk + br) * K + k0 + sSeg * 8,
                     &Bs[bi][br * 32 + sSeg * 8]);
        }
    };

    v8f acc[2][4];
    #pragma unroll
    for (int mt = 0; mt < 2; ++mt)
        #pragma unroll
        for (int t = 0; t < 4; ++t) acc[mt][t] = (v8f){};

    int buf = 0;
    stage(0, 0);
    wait_async();
    __syncthreads();

    for (int k0 = 0; k0 < K; k0 += 32) {
        if (k0 + 32 < K) stage(buf ^ 1, k0 + 32);

        // fragments from LDS (ds_load_b128 pairs)
        v16bf af0 = load_frag(&As[buf][(mw * 32 +  0 + r16) * 32 + 8 * g]);
        v16bf af1 = load_frag(&As[buf][(mw * 32 + 16 + r16) * 32 + 8 * g]);
        #pragma unroll
        for (int t = 0; t < 4; ++t) {
            v16bf bf = load_frag(&Bs[buf][(nw * 64 + t * 16 + r16) * 32 + 8 * g]);
            acc[0][t] = wmma_bf16(af0, bf, acc[0][t]);
            acc[1][t] = wmma_bf16(af1, bf, acc[1][t]);
        }

        wait_async();        // copies for buf^1 have landed
        __syncthreads();     // all waves done reading buf; buf^1 visible
        buf ^= 1;
    }

    // Epilogue: C layout => VGPR r holds row (tile + r + 8g), col n = tile + r16
    #pragma unroll
    for (int mt = 0; mt < 2; ++mt) {
        #pragma unroll
        for (int t = 0; t < 4; ++t) {
            int col = n0 + t * 16 + r16;
            float bs = bias[col];
            #pragma unroll
            for (int r = 0; r < 8; ++r) {
                int row = m0 + mt * 16 + r + 8 * g;
                float v = acc[mt][t][r] + bs;
                if (OUT_BF16)
                    ((ushort_t*)Cout)[(size_t)row * N + col] = f2bf_bits(v);
                else
                    ((float*)Cout)[(size_t)row * N + col] = v;
            }
        }
    }
}

// ---------------- RoPE on last 64 dims of each head (in-place, bf16) ----------------
__global__ void rope_kernel(ushort_t* __restrict__ qk,
                            const float* __restrict__ fcos, const float* __restrict__ fsin) {
    long idx = (long)blockIdx.x * blockDim.x + threadIdx.x;
    long total = (long)MTOT * H_ * (ROPE_ / 2);
    if (idx >= total) return;
    int i = (int)(idx & 31);            // pair index within rope dims
    int h = (int)((idx >> 5) & 15);
    int m = (int)(idx >> 9);            // flattened b*S + s
    int s = m & (S_ - 1);
    long base = (long)m * NQK + h * QKD_ + NOPE_ + 2 * i;
    float c  = fcos[s * (ROPE_ / 2) + i];
    float sn = fsin[s * (ROPE_ / 2) + i];
    float xr = bf2f(qk[base]);
    float xi = bf2f(qk[base + 1]);
    qk[base]     = f2bf_bits(xr * c - xi * sn);
    qk[base + 1] = f2bf_bits(xr * sn + xi * c);
}

// ---------------- Flash attention: one wave per (b, h, 16-row q tile) ----------------
__global__ void __launch_bounds__(32)
attn_kernel(const ushort_t* __restrict__ qb, const ushort_t* __restrict__ kb,
            const ushort_t* __restrict__ vb, ushort_t* __restrict__ ob) {
    __shared__ __align__(16) ushort_t pl[16 * 32];   // P staging: 16 q-rows x 32 keys

    const int lane = threadIdx.x & 31;
    const int g    = lane >> 4;
    const int r16  = lane & 15;
    const int q0   = blockIdx.x * 16;
    const int h    = blockIdx.y;
    const int b    = blockIdx.z;
    const float scale = 0.07216878364870322f;        // 192^-0.5

    // Preload 16 x 192 Q tile as 6 A-fragments
    v16bf qf[6];
    {
        const ushort_t* qrow = qb + (size_t)(b * S_ + q0 + r16) * NQK + h * QKD_ + 8 * g;
        #pragma unroll
        for (int kk = 0; kk < 6; ++kk) qf[kk] = load_frag(qrow + kk * 32);
    }

    float mrow[8], lrow[8];
    v8f oacc[8];
    #pragma unroll
    for (int r = 0; r < 8; ++r) { mrow[r] = -1e30f; lrow[r] = 0.0f; }
    #pragma unroll
    for (int t = 0; t < 8; ++t) oacc[t] = (v8f){};

    const int ktMax = (q0 + 15) >> 5;                // causal: skip fully-masked blocks
    for (int kt = 0; kt <= ktMax; ++kt) {
        const int k0 = kt * 32;

        // --- scores: 16 q-rows x 32 keys ---
        v8f s0 = {}, s1 = {};
        const ushort_t* krow0 = kb + (size_t)(b * S_ + k0 + r16) * NQK + h * QKD_ + 8 * g;
        const ushort_t* krow1 = krow0 + (size_t)16 * NQK;
        #pragma unroll
        for (int kk = 0; kk < 6; ++kk) {
            v16bf kf0 = load_frag(krow0 + kk * 32);
            v16bf kf1 = load_frag(krow1 + kk * 32);
            s0 = wmma_bf16(qf[kk], kf0, s0);
            s1 = wmma_bf16(qf[kk], kf1, s1);
        }

        // --- online softmax (C layout: row r+8g lives in VGPR r of half-wave g) ---
        float p0v[8], p1v[8], alpha[8];
        #pragma unroll
        for (int r = 0; r < 8; ++r) {
            int rowq = q0 + r + 8 * g;
            float v0 = s0[r] * scale + ((k0 + r16)      <= rowq ? 0.0f : -1.0e9f);
            float v1 = s1[r] * scale + ((k0 + 16 + r16) <= rowq ? 0.0f : -1.0e9f);
            float tmax = fmaxf(v0, v1);
            #pragma unroll
            for (int d = 1; d < 16; d <<= 1) tmax = fmaxf(tmax, __shfl_xor(tmax, d, 16));
            float mnew = fmaxf(mrow[r], tmax);
            float a  = __expf(mrow[r] - mnew);
            float p0 = __expf(v0 - mnew);
            float p1 = __expf(v1 - mnew);
            float ps = p0 + p1;
            #pragma unroll
            for (int d = 1; d < 16; d <<= 1) ps += __shfl_xor(ps, d, 16);
            lrow[r]  = lrow[r] * a + ps;
            mrow[r]  = mnew;
            alpha[r] = a;
            p0v[r] = p0; p1v[r] = p1;
        }
        #pragma unroll
        for (int t = 0; t < 8; ++t)
            #pragma unroll
            for (int r = 0; r < 8; ++r) oacc[t][r] *= alpha[r];

        // --- transpose P (C layout -> A layout) through LDS ---
        __syncthreads();
        #pragma unroll
        for (int r = 0; r < 8; ++r) {
            pl[(r + 8 * g) * 32 + r16]      = f2bf_bits(p0v[r]);
            pl[(r + 8 * g) * 32 + 16 + r16] = f2bf_bits(p1v[r]);
        }
        __syncthreads();
        v16bf pf = load_frag(&pl[r16 * 32 + 8 * g]);

        // --- oacc += P(16x32) x V(32x128), 8 N-tiles ---
        const ushort_t* vbase = vb + (size_t)(b * S_ + k0) * NV + h * VD_;
        #pragma unroll
        for (int t = 0; t < 8; ++t) {
            v16bf vf;
            #pragma unroll
            for (int e = 0; e < 16; ++e) {
                int key = (e < 8) ? (8 * g + e) : (16 + 8 * g + (e - 8));
                vf[e] = bits2bf(vbase[(size_t)key * NV + t * 16 + r16]);
            }
            oacc[t] = wmma_bf16(pf, vf, oacc[t]);
        }
    }

    // --- normalize & store (bf16) ---
    #pragma unroll
    for (int t = 0; t < 8; ++t)
        #pragma unroll
        for (int r = 0; r < 8; ++r) {
            int row = q0 + r + 8 * g;
            float v = oacc[t][r] / lrow[r];
            ob[(size_t)(b * S_ + row) * NV + h * VD_ + t * 16 + r16] = f2bf_bits(v);
        }
}

// ---------------- host-side orchestration ----------------
extern "C" void kernel_launch(void* const* d_in, const int* in_sizes, int n_in,
                              void* d_out, int out_size, void* d_ws, size_t ws_size,
                              hipStream_t stream) {
    const float* x    = (const float*)d_in[0];
    const float* wq_w = (const float*)d_in[1];
    const float* wq_b = (const float*)d_in[2];
    const float* wk_w = (const float*)d_in[3];
    const float* wk_b = (const float*)d_in[4];
    const float* wv_w = (const float*)d_in[5];
    const float* wv_b = (const float*)d_in[6];
    const float* wo_w = (const float*)d_in[7];
    const float* wo_b = (const float*)d_in[8];
    const float* fcos = (const float*)d_in[9];
    const float* fsin = (const float*)d_in[10];
    // d_in[11] mask (causal mask computed analytically), d_in[12] start_pos (==0): unused

    char* ws = (char*)d_ws;
    size_t off = 0;
    auto alloc = [&](size_t bytes) {
        char* p = ws + off;
        off += (bytes + 255) & ~(size_t)255;
        return p;
    };
    ushort_t* xb   = (ushort_t*)alloc((size_t)MTOT * D_ * 2);  // x, bf16
    ushort_t* wqT  = (ushort_t*)alloc((size_t)NQK * D_ * 2);   // [3072][2048]
    ushort_t* wkT  = (ushort_t*)alloc((size_t)NQK * D_ * 2);
    ushort_t* wvT  = (ushort_t*)alloc((size_t)NV  * D_ * 2);   // [2048][2048]
    ushort_t* woT  = (ushort_t*)alloc((size_t)D_  * NV  * 2);  // [2048][2048]
    ushort_t* qbuf = (ushort_t*)alloc((size_t)MTOT * NQK * 2);
    ushort_t* kbuf = (ushort_t*)alloc((size_t)MTOT * NQK * 2);
    ushort_t* vbuf = (ushort_t*)alloc((size_t)MTOT * NV  * 2);
    ushort_t* obuf = (ushort_t*)alloc((size_t)MTOT * NV  * 2);

    // 1. convert activations + weights to bf16 (weights transposed to [N][K])
    long nx = (long)MTOT * D_;
    conv_bf16<<<(unsigned)((nx + 255) / 256), 256, 0, stream>>>(x, xb, nx);
    long nwq = (long)D_ * NQK;
    conv_transpose<<<(unsigned)((nwq + 255) / 256), 256, 0, stream>>>(wq_w, wqT, D_, NQK);
    conv_transpose<<<(unsigned)((nwq + 255) / 256), 256, 0, stream>>>(wk_w, wkT, D_, NQK);
    long nwv = (long)D_ * NV;
    conv_transpose<<<(unsigned)((nwv + 255) / 256), 256, 0, stream>>>(wv_w, wvT, D_, NV);
    conv_transpose<<<(unsigned)((nwv + 255) / 256), 256, 0, stream>>>(wo_w, woT, NV, D_);

    // 2. QKV projections (bf16 out, bias fused)
    gemm_wmma<true><<<dim3(NQK / 256, MTOT / 64), 256, 0, stream>>>(xb, wqT, wq_b, qbuf, MTOT, NQK, D_);
    gemm_wmma<true><<<dim3(NQK / 256, MTOT / 64), 256, 0, stream>>>(xb, wkT, wk_b, kbuf, MTOT, NQK, D_);
    gemm_wmma<true><<<dim3(NV  / 256, MTOT / 64), 256, 0, stream>>>(xb, wvT, wv_b, vbuf, MTOT, NV,  D_);

    // 3. RoPE on q, k
    long nr = (long)MTOT * H_ * (ROPE_ / 2);
    rope_kernel<<<(unsigned)((nr + 255) / 256), 256, 0, stream>>>(qbuf, fcos, fsin);
    rope_kernel<<<(unsigned)((nr + 255) / 256), 256, 0, stream>>>(kbuf, fcos, fsin);

    // 4. flash attention (causal)
    attn_kernel<<<dim3(S_ / 16, H_, B_), 32, 0, stream>>>(qbuf, kbuf, vbuf, obuf);

    // 5. output projection (f32 out, bias fused)
    gemm_wmma<false><<<dim3(D_ / 256, MTOT / 64), 256, 0, stream>>>(obuf, woT, wo_b, d_out, MTOT, D_, D_);

    (void)in_sizes; (void)n_in; (void)out_size; (void)ws_size;
}
